// HierarchicalCrystalTransformer_72121090834920
// MI455X (gfx1250) — compile-verified, matrix operand order checked by reading, not run on hardware
//
#include <hip/hip_runtime.h>

typedef unsigned short u16;
typedef __attribute__((ext_vector_type(16))) __bf16 v16bf;
typedef __attribute__((ext_vector_type(8)))  float  v8f;

#define S_LEN 1024
#define D_MOD 768
#define N_HEAD 12
#define HD 64
#define DFF 3072
#define NVOC 2000
#define NTOK 4096   /* B*S */
#define TS 40       /* flash LDS row stride (u16) for 32-wide tiles */
#define TSW 72      /* gemm LDS row stride (u16) for 64-wide K tiles, 144B */

union Frag { uint4 q[2]; v16bf v; };

__device__ __forceinline__ u16 f2bf(float f) {
  unsigned u = __float_as_uint(f);
  u += 0x7FFFu + ((u >> 16) & 1u);
  return (u16)(u >> 16);
}

__device__ __forceinline__ v8f wmma_bf16(Frag a, Frag b, v8f c) {
  return __builtin_amdgcn_wmma_f32_16x16x32_bf16(false, a.v, false, b.v,
                                                 (short)0, c, false, false);
}

// ---------------------------------------------------------------------------
// Weight convert + transpose: W[K,N] fp32 -> Wt[N,K] bf16 (32x32 LDS tiles)
// grid (N/32, K/32), 256 threads
// ---------------------------------------------------------------------------
__global__ __launch_bounds__(256) void k_wt(
    const float* __restrict__ W, u16* __restrict__ Wt, int K, int N)
{
  __shared__ float tile[32][33];
  int n0 = blockIdx.x * 32, k0 = blockIdx.y * 32;
  int kl = threadIdx.x >> 3, ng = threadIdx.x & 7;
  float4 f = *(const float4*)&W[(size_t)(k0 + kl) * N + n0 + 4 * ng];
  tile[kl][4 * ng + 0] = f.x;
  tile[kl][4 * ng + 1] = f.y;
  tile[kl][4 * ng + 2] = f.z;
  tile[kl][4 * ng + 3] = f.w;
  __syncthreads();
  int nl = threadIdx.x >> 3, kg = threadIdx.x & 7;
  union { u16 h[4]; uint2 q; } o;
  #pragma unroll
  for (int j = 0; j < 4; j++) o.h[j] = f2bf(tile[4 * kg + j][nl]);
  *(uint2*)&Wt[(size_t)(n0 + nl) * K + k0 + 4 * kg] = o.q;
}

// ---------------------------------------------------------------------------
// Elementwise fp32 -> bf16 (lm_head, already [N,K]); n must be multiple of 4
// ---------------------------------------------------------------------------
__global__ __launch_bounds__(256) void k_wc(
    const float* __restrict__ in, u16* __restrict__ out, int n)
{
  int i = (blockIdx.x * 256 + threadIdx.x) * 4;
  if (i >= n) return;
  float4 f = *(const float4*)&in[i];
  union { u16 h[4]; uint2 q; } o;
  o.h[0] = f2bf(f.x); o.h[1] = f2bf(f.y); o.h[2] = f2bf(f.z); o.h[3] = f2bf(f.w);
  *(uint2*)&out[i] = o.q;
}

// ---------------------------------------------------------------------------
// Embedding gather + LayerNorm -> fp32 x
// ---------------------------------------------------------------------------
__global__ __launch_bounds__(256) void k_embed(
    const int* __restrict__ ids, const int* __restrict__ segs,
    const float* __restrict__ tok, const float* __restrict__ pos,
    const float* __restrict__ typ, const float* __restrict__ g,
    const float* __restrict__ b, float* __restrict__ x)
{
  int row = blockIdx.x;
  int s   = row & (S_LEN - 1);
  int id  = ids[row];  id = id < 0 ? 0 : (id > NVOC - 1 ? NVOC - 1 : id);
  int sg  = segs[row]; sg = sg < 0 ? 0 : (sg > 3 ? 3 : sg);

  float v[3], sum = 0.f, sum2 = 0.f;
  #pragma unroll
  for (int i = 0; i < 3; i++) {
    int d = threadIdx.x + 256 * i;
    float t = tok[(size_t)id * D_MOD + d] + pos[(size_t)s * D_MOD + d]
            + typ[(size_t)sg * D_MOD + d];
    v[i] = t; sum += t; sum2 += t * t;
  }
  __shared__ float r1[256], r2[256];
  r1[threadIdx.x] = sum; r2[threadIdx.x] = sum2; __syncthreads();
  for (int st = 128; st > 0; st >>= 1) {
    if (threadIdx.x < st) { r1[threadIdx.x] += r1[threadIdx.x + st];
                            r2[threadIdx.x] += r2[threadIdx.x + st]; }
    __syncthreads();
  }
  float mean = r1[0] * (1.f / D_MOD);
  float var  = r2[0] * (1.f / D_MOD) - mean * mean;
  float rstd = rsqrtf(var + 1e-12f);
  #pragma unroll
  for (int i = 0; i < 3; i++) {
    int d = threadIdx.x + 256 * i;
    x[(size_t)row * D_MOD + d] = (v[i] - mean) * rstd * g[d] + b[d];
  }
}

// ---------------------------------------------------------------------------
// LayerNorm fp32 -> bf16
// ---------------------------------------------------------------------------
__global__ __launch_bounds__(256) void k_ln(
    const float* __restrict__ x, const float* __restrict__ g,
    const float* __restrict__ b, u16* __restrict__ y)
{
  int row = blockIdx.x;
  const float* xr = x + (size_t)row * D_MOD;
  float v[3], sum = 0.f, sum2 = 0.f;
  #pragma unroll
  for (int i = 0; i < 3; i++) {
    float t = xr[threadIdx.x + 256 * i];
    v[i] = t; sum += t; sum2 += t * t;
  }
  __shared__ float r1[256], r2[256];
  r1[threadIdx.x] = sum; r2[threadIdx.x] = sum2; __syncthreads();
  for (int st = 128; st > 0; st >>= 1) {
    if (threadIdx.x < st) { r1[threadIdx.x] += r1[threadIdx.x + st];
                            r2[threadIdx.x] += r2[threadIdx.x + st]; }
    __syncthreads();
  }
  float mean = r1[0] * (1.f / D_MOD);
  float var  = r2[0] * (1.f / D_MOD) - mean * mean;
  float rstd = rsqrtf(var + 1e-12f);
  #pragma unroll
  for (int i = 0; i < 3; i++) {
    int d = threadIdx.x + 256 * i;
    y[(size_t)row * D_MOD + d] = f2bf((v[i] - mean) * rstd * g[d] + b[d]);
  }
}

// ---------------------------------------------------------------------------
// Tiled WMMA GEMM:  C[M,N] = act(A_bf16[M,K] @ Wt_bf16[N,K]^T + bias) (+ res)
// block = 128x128 tile, 256 thr = 8 waves (4x2); per wave 2 M-tiles x 4 N-tiles.
// BK=64: 16 WMMAs per barrier pair; staging is pure b128 copies (pre-converted
// bf16 weights). N-edge via clamped B rows (always-valid loads).
// outb modes: 0 row-major, 1 per-head [b,h,s,hd] (Q/K), 2 per-head-T [b,h,hd,s] (V)
// ---------------------------------------------------------------------------
__global__ __launch_bounds__(256) void k_gemm(
    const u16* __restrict__ A, const u16* __restrict__ Wt,
    const float* __restrict__ bias, const float* __restrict__ res,
    float* __restrict__ outf, u16* __restrict__ outb,
    int M, int N, int K, int mode, int act)
{
  __shared__ __align__(16) u16 Ash[128 * TSW];
  __shared__ __align__(16) u16 Bsh[128 * TSW];

  int t = threadIdx.x;
  int nB = blockIdx.x * 128, mB = blockIdx.y * 128;
  int lane = t & 31, wv = t >> 5;
  int half = lane >> 4, l15 = lane & 15;
  int mw = wv & 3, nw = wv >> 2;   // 4 x 2 wave grid

  v8f acc[2][4];
  #pragma unroll
  for (int i = 0; i < 2; i++)
    #pragma unroll
    for (int j = 0; j < 4; j++)
      #pragma unroll
      for (int r = 0; r < 8; r++) acc[i][j][r] = 0.f;

  int row = t >> 1, h32 = t & 1;          // staging coords (fixed per thread)
  int nc = nB + row; if (nc > N - 1) nc = N - 1;  // clamp for N-edge tiles

  for (int k0 = 0; k0 < K; k0 += 64) {
    { // stage A: 128x64 bf16, 64B per thread
      const uint4* sa = (const uint4*)&A[(size_t)(mB + row) * K + k0 + 32 * h32];
      uint4* da = (uint4*)&Ash[row * TSW + 32 * h32];
      da[0] = sa[0]; da[1] = sa[1]; da[2] = sa[2]; da[3] = sa[3];
    }
    { // stage B: Wt[N,K] bf16, straight copy
      const uint4* sb = (const uint4*)&Wt[(size_t)nc * K + k0 + 32 * h32];
      uint4* db = (uint4*)&Bsh[row * TSW + 32 * h32];
      db[0] = sb[0]; db[1] = sb[1]; db[2] = sb[2]; db[3] = sb[3];
    }
    __syncthreads();

    #pragma unroll
    for (int kc = 0; kc < 2; kc++) {
      int co = 32 * kc + 8 * half;
      Frag a0, a1, bfr[4];
      int ra = (mw * 32 + l15) * TSW + co;
      a0.q[0] = *(const uint4*)&Ash[ra];
      a0.q[1] = *(const uint4*)&Ash[ra + 16];
      a1.q[0] = *(const uint4*)&Ash[ra + 16 * TSW];
      a1.q[1] = *(const uint4*)&Ash[ra + 16 * TSW + 16];
      #pragma unroll
      for (int j = 0; j < 4; j++) {
        int rb = (nw * 64 + 16 * j + l15) * TSW + co;
        bfr[j].q[0] = *(const uint4*)&Bsh[rb];
        bfr[j].q[1] = *(const uint4*)&Bsh[rb + 16];
      }
      #pragma unroll
      for (int j = 0; j < 4; j++) {
        acc[0][j] = wmma_bf16(a0, bfr[j], acc[0][j]);
        acc[1][j] = wmma_bf16(a1, bfr[j], acc[1][j]);
      }
    }
    __syncthreads();
  }

  #pragma unroll
  for (int j = 0; j < 4; j++) {
    int n = nB + nw * 64 + 16 * j + l15;
    bool nok = n < N;
    float bv = (bias && nok) ? bias[n] : 0.f;
    #pragma unroll
    for (int i = 0; i < 2; i++) {
      #pragma unroll
      for (int r = 0; r < 8; r++) {
        int m = mB + mw * 32 + 16 * i + r + 8 * half;
        float v = acc[i][j][r] + bv;
        if (act) v = 0.5f * v * (1.f + erff(v * 0.70710678118f));
        if (!nok) continue;
        if (res)  v += res[(size_t)m * N + n];
        if (outf) outf[(size_t)m * N + n] = v;
        if (outb) {
          u16 hv = f2bf(v);
          if (mode == 0) outb[(size_t)m * N + n] = hv;
          else {
            int bI = m >> 10, sI = m & (S_LEN - 1), hI = n >> 6, dI = n & 63;
            size_t idx = (mode == 1)
              ? ((size_t)(bI * N_HEAD + hI) * S_LEN + sI) * HD + dI
              : ((size_t)(bI * N_HEAD + hI) * HD + dI) * S_LEN + sI;
            outb[idx] = hv;
          }
        }
      }
    }
  }
}

// ---------------------------------------------------------------------------
// Flash attention (wave-per-16-query-rows), bf16 WMMA for QK^T and P*V.
// q,k: [B*H, S, HD] bf16;  vt: [B*H, HD, S] bf16;  ctx out: [B*S, D] bf16
// grid (B*H, S/128), 256 threads (8 waves)
// ---------------------------------------------------------------------------
__global__ __launch_bounds__(256) void k_flash(
    const u16* __restrict__ qb, const u16* __restrict__ kb,
    const u16* __restrict__ vtb, u16* __restrict__ ctxb, int causal)
{
  __shared__ __align__(16) u16 pl[8 * 16 * TS]; // per-wave 16x32 prob tile

  int t = threadIdx.x, lane = t & 31, wv = t >> 5;
  int half = lane >> 4, l15 = lane & 15;
  int bh = blockIdx.x;
  int qbase = blockIdx.y * 128 + wv * 16;
  size_t head = (size_t)bh * S_LEN * HD;

  Frag q0, q1;
  {
    const u16* qp = &qb[head + (size_t)(qbase + l15) * HD];
    q0.q[0] = *(const uint4*)&qp[8 * half];
    q0.q[1] = *(const uint4*)&qp[8 * half + 16];
    q1.q[0] = *(const uint4*)&qp[32 + 8 * half];
    q1.q[1] = *(const uint4*)&qp[48 + 8 * half];
  }

  v8f o[4];
  float mrow[8], lrow[8];
  #pragma unroll
  for (int c = 0; c < 4; c++)
    #pragma unroll
    for (int r = 0; r < 8; r++) o[c][r] = 0.f;
  #pragma unroll
  for (int r = 0; r < 8; r++) { mrow[r] = -1e30f; lrow[r] = 0.f; }

  int nkt = causal ? ((qbase + 16 + 31) >> 5) : (S_LEN >> 5);
  for (int kt = 0; kt < nkt; kt++) {
    int k0 = kt * 32;
    v8f s0 = {}, s1 = {};
    {
      Frag kf;
      const u16* kp = &kb[head + (size_t)(k0 + l15) * HD];
      kf.q[0] = *(const uint4*)&kp[8 * half];
      kf.q[1] = *(const uint4*)&kp[8 * half + 16];
      s0 = wmma_bf16(q0, kf, s0);
      kf.q[0] = *(const uint4*)&kp[32 + 8 * half];
      kf.q[1] = *(const uint4*)&kp[48 + 8 * half];
      s0 = wmma_bf16(q1, kf, s0);
      const u16* kp2 = kp + 16 * HD;
      kf.q[0] = *(const uint4*)&kp2[8 * half];
      kf.q[1] = *(const uint4*)&kp2[8 * half + 16];
      s1 = wmma_bf16(q0, kf, s1);
      kf.q[0] = *(const uint4*)&kp2[32 + 8 * half];
      kf.q[1] = *(const uint4*)&kp2[48 + 8 * half];
      s1 = wmma_bf16(q1, kf, s1);
    }

    #pragma unroll
    for (int r = 0; r < 8; r++) {
      int qpos = qbase + r + 8 * half;
      float a  = s0[r] * 0.125f;   // 1/sqrt(64)
      float b2 = s1[r] * 0.125f;
      if (causal) {
        if (k0 + l15 > qpos)      a  = -1e30f;
        if (k0 + 16 + l15 > qpos) b2 = -1e30f;
      }
      float mx = fmaxf(a, b2);
      mx = fmaxf(mx, __shfl_xor(mx, 1, 16));
      mx = fmaxf(mx, __shfl_xor(mx, 2, 16));
      mx = fmaxf(mx, __shfl_xor(mx, 4, 16));
      mx = fmaxf(mx, __shfl_xor(mx, 8, 16));
      float mnew = fmaxf(mrow[r], mx);
      float sc   = __expf(mrow[r] - mnew);
      mrow[r] = mnew;
      float p0 = __expf(a - mnew), p1 = __expf(b2 - mnew);
      float ps = p0 + p1;
      ps += __shfl_xor(ps, 1, 16);
      ps += __shfl_xor(ps, 2, 16);
      ps += __shfl_xor(ps, 4, 16);
      ps += __shfl_xor(ps, 8, 16);
      lrow[r] = lrow[r] * sc + ps;
      #pragma unroll
      for (int c = 0; c < 4; c++) o[c][r] *= sc;
      int mr = r + 8 * half;
      pl[wv * 16 * TS + mr * TS + l15]      = f2bf(p0);
      pl[wv * 16 * TS + mr * TS + 16 + l15] = f2bf(p1);
    }
    // wave-private LDS RAW across lanes; waves diverge so no block barrier
    asm volatile("s_wait_dscnt 0x0" ::: "memory");

    Frag pf;
    pf.q[0] = *(const uint4*)&pl[wv * 16 * TS + l15 * TS + 8 * half];
    pf.q[1] = *(const uint4*)&pl[wv * 16 * TS + l15 * TS + 8 * half + 16];

    #pragma unroll
    for (int c = 0; c < 4; c++) {
      Frag vf;
      const u16* vp = &vtb[head + (size_t)(16 * c + l15) * S_LEN + k0 + 8 * half];
      vf.q[0] = *(const uint4*)vp;
      vf.q[1] = *(const uint4*)(vp + 16);
      o[c] = wmma_bf16(pf, vf, o[c]);
    }
  }

  int bI = bh / N_HEAD, hI = bh % N_HEAD;
  #pragma unroll
  for (int r = 0; r < 8; r++) {
    float inv = 1.f / lrow[r];
    int srow = qbase + r + 8 * half;
    u16* cp = &ctxb[((size_t)bI * S_LEN + srow) * D_MOD + hI * HD];
    #pragma unroll
    for (int c = 0; c < 4; c++)
      cp[16 * c + l15] = f2bf(o[c][r] * inv);
  }
}

// ===========================================================================
// Host side
// ===========================================================================
struct Lin { const float *w, *b; };
struct Mha { Lin q, k, v, o; };
struct LnP { const float *g, *b; };
struct TlP { Mha a; const float *b1, *b2; LnP ln1, ln2; const float *w1, *w2; };
struct ClP { Mha a; LnP ln; };

static Mha mha_at(void* const* d, int i) {
  Mha m;
  m.k.b = (const float*)d[i + 0]; m.k.w = (const float*)d[i + 1];
  m.o.b = (const float*)d[i + 2]; m.o.w = (const float*)d[i + 3];
  m.q.b = (const float*)d[i + 4]; m.q.w = (const float*)d[i + 5];
  m.v.b = (const float*)d[i + 6]; m.v.w = (const float*)d[i + 7];
  return m;
}
static TlP tl_at(void* const* d, int i) {
  TlP t; t.a = mha_at(d, i);
  t.b1 = (const float*)d[i + 8];  t.b2 = (const float*)d[i + 9];
  t.ln1.b = (const float*)d[i + 10]; t.ln1.g = (const float*)d[i + 11];
  t.ln2.b = (const float*)d[i + 12]; t.ln2.g = (const float*)d[i + 13];
  t.w1 = (const float*)d[i + 14]; t.w2 = (const float*)d[i + 15];
  return t;
}
static ClP cl_at(void* const* d, int i) {
  ClP c; c.a = mha_at(d, i);
  c.ln.b = (const float*)d[i + 8]; c.ln.g = (const float*)d[i + 9];
  return c;
}

// convert W[K,N] fp32 -> wscr[N,K] bf16, then GEMM against it
static void gemmW(hipStream_t st, const u16* A, const float* Wf, const float* bias,
                  const float* res, float* outf, u16* outb,
                  int M, int N, int K, int mode, int act, u16* wscr) {
  k_wt<<<dim3(N / 32, K / 32), 256, 0, st>>>(Wf, wscr, K, N);
  dim3 g((N + 127) / 128, M / 128), b(256);
  k_gemm<<<g, b, 0, st>>>(A, wscr, bias, res, outf, outb, M, N, K, mode, act);
}

static void run_tl(hipStream_t st, const TlP& p, float* x, u16* yb, u16* qb,
                   u16* kbuf, u16* vtb, u16* ctxb, u16* f1b, u16* snap, u16* wscr) {
  k_ln<<<NTOK, 256, 0, st>>>(x, p.ln1.g, p.ln1.b, yb);
  gemmW(st, yb, p.a.q.w, p.a.q.b, nullptr, nullptr, qb,   NTOK, D_MOD, D_MOD, 1, 0, wscr);
  gemmW(st, yb, p.a.k.w, p.a.k.b, nullptr, nullptr, kbuf, NTOK, D_MOD, D_MOD, 1, 0, wscr);
  gemmW(st, yb, p.a.v.w, p.a.v.b, nullptr, nullptr, vtb,  NTOK, D_MOD, D_MOD, 2, 0, wscr);
  k_flash<<<dim3(4 * N_HEAD, S_LEN / 128), 256, 0, st>>>(qb, kbuf, vtb, ctxb, 1);
  gemmW(st, ctxb, p.a.o.w, p.a.o.b, x, x, nullptr, NTOK, D_MOD, D_MOD, 0, 0, wscr);
  k_ln<<<NTOK, 256, 0, st>>>(x, p.ln2.g, p.ln2.b, yb);
  gemmW(st, yb,  p.w1, p.b1, nullptr, nullptr, f1b, NTOK, DFF,   D_MOD, 0, 1, wscr);
  gemmW(st, f1b, p.w2, p.b2, x, x, snap,            NTOK, D_MOD, DFF,   0, 0, wscr);
}

static void run_cl(hipStream_t st, const ClP& p, float* x, const u16* ctx_src,
                   u16* yb, u16* qb, u16* kbuf, u16* vtb, u16* ctxb, u16* snap,
                   u16* wscr) {
  k_ln<<<NTOK, 256, 0, st>>>(x, p.ln.g, p.ln.b, yb);
  gemmW(st, yb,      p.a.q.w, p.a.q.b, nullptr, nullptr, qb,   NTOK, D_MOD, D_MOD, 1, 0, wscr);
  gemmW(st, ctx_src, p.a.k.w, p.a.k.b, nullptr, nullptr, kbuf, NTOK, D_MOD, D_MOD, 1, 0, wscr);
  gemmW(st, ctx_src, p.a.v.w, p.a.v.b, nullptr, nullptr, vtb,  NTOK, D_MOD, D_MOD, 2, 0, wscr);
  k_flash<<<dim3(4 * N_HEAD, S_LEN / 128), 256, 0, st>>>(qb, kbuf, vtb, ctxb, 0);
  gemmW(st, ctxb, p.a.o.w, p.a.o.b, x, x, snap, NTOK, D_MOD, D_MOD, 0, 0, wscr);
}

extern "C" void kernel_launch(void* const* d_in, const int* in_sizes, int n_in,
                              void* d_out, int out_size, void* d_ws, size_t ws_size,
                              hipStream_t stream) {
  (void)in_sizes; (void)n_in; (void)out_size; (void)ws_size;
  const int* ids  = (const int*)d_in[0];
  const int* segs = (const int*)d_in[1];
  // d_in[2] = attention_mask (all ones -> no-op in reference semantics)

  // params leaves, jax tree-flatten order (dict keys sorted, lists in order), base=3
  TlP atom0 = tl_at(d_in, 3),  atom1 = tl_at(d_in, 19), atom2 = tl_at(d_in, 35);
  ClP atomx = cl_at(d_in, 51);
  TlP comp0 = tl_at(d_in, 61), comp1 = tl_at(d_in, 77);
  const float* emb_b = (const float*)d_in[93];
  const float* emb_g = (const float*)d_in[94];
  TlP int0 = tl_at(d_in, 95),  int1 = tl_at(d_in, 111);
  ClP latx = cl_at(d_in, 127);
  TlP lat0 = tl_at(d_in, 137), lat1 = tl_at(d_in, 153);
  const float* lm  = (const float*)d_in[169];   // [V, D]
  const float* pos = (const float*)d_in[170];
  ClP sgx = cl_at(d_in, 171);
  TlP sg0 = tl_at(d_in, 181),  sg1 = tl_at(d_in, 197);
  const float* tok = (const float*)d_in[213];
  const float* typ = (const float*)d_in[214];

  // workspace carve-up (~94.5 MiB)
  char* wsp = (char*)d_ws; size_t off = 0;
  auto alloc = [&](size_t bytes) -> void* {
    void* p = wsp + off; off += (bytes + 255) & ~(size_t)255; return p;
  };
  float* x    = (float*)alloc((size_t)NTOK * D_MOD * 4);
  u16* yb     = (u16*)alloc((size_t)NTOK * D_MOD * 2);
  u16* qb     = (u16*)alloc((size_t)NTOK * D_MOD * 2);
  u16* kbuf   = (u16*)alloc((size_t)NTOK * D_MOD * 2);
  u16* vtb    = (u16*)alloc((size_t)NTOK * D_MOD * 2);
  u16* ctxb   = (u16*)alloc((size_t)NTOK * D_MOD * 2);
  u16* f1b    = (u16*)alloc((size_t)NTOK * DFF   * 2);
  u16* comp_b = (u16*)alloc((size_t)NTOK * D_MOD * 2);
  u16* sg_b   = (u16*)alloc((size_t)NTOK * D_MOD * 2);
  u16* lat_b  = (u16*)alloc((size_t)NTOK * D_MOD * 2);
  u16* xfb    = (u16*)alloc((size_t)NTOK * D_MOD * 2);
  u16* wscr   = (u16*)alloc((size_t)DFF  * D_MOD * 2);  // bf16 weight scratch

  k_embed<<<NTOK, 256, 0, stream>>>(ids, segs, tok, pos, typ, emb_g, emb_b, x);

  run_tl(stream, comp0, x, yb, qb, kbuf, vtb, ctxb, f1b, nullptr, wscr);
  run_tl(stream, comp1, x, yb, qb, kbuf, vtb, ctxb, f1b, comp_b, wscr);

  run_tl(stream, sg0, x, yb, qb, kbuf, vtb, ctxb, f1b, nullptr, wscr);
  run_tl(stream, sg1, x, yb, qb, kbuf, vtb, ctxb, f1b, nullptr, wscr);
  run_cl(stream, sgx, x, comp_b, yb, qb, kbuf, vtb, ctxb, sg_b, wscr);

  run_tl(stream, lat0, x, yb, qb, kbuf, vtb, ctxb, f1b, nullptr, wscr);
  run_tl(stream, lat1, x, yb, qb, kbuf, vtb, ctxb, f1b, nullptr, wscr);
  run_cl(stream, latx, x, sg_b, yb, qb, kbuf, vtb, ctxb, lat_b, wscr);

  run_tl(stream, atom0, x, yb, qb, kbuf, vtb, ctxb, f1b, nullptr, wscr);
  run_tl(stream, atom1, x, yb, qb, kbuf, vtb, ctxb, f1b, nullptr, wscr);
  run_tl(stream, atom2, x, yb, qb, kbuf, vtb, ctxb, f1b, nullptr, wscr);
  run_cl(stream, atomx, x, lat_b, yb, qb, kbuf, vtb, ctxb, nullptr, wscr);

  run_tl(stream, int0, x, yb, qb, kbuf, vtb, ctxb, f1b, nullptr, wscr);
  run_tl(stream, int1, x, yb, qb, kbuf, vtb, ctxb, f1b, xfb, wscr);

  // lm_head is [V,D] == [N,K] already: elementwise convert then GEMM (N=2000 edge)
  k_wc<<<(NVOC * D_MOD / 4 + 255) / 256, 256, 0, stream>>>(lm, wscr, NVOC * D_MOD);
  {
    dim3 g((NVOC + 127) / 128, NTOK / 128), b(256);
    k_gemm<<<g, b, 0, stream>>>(xfb, wscr, nullptr, nullptr, (float*)d_out,
                                nullptr, NTOK, NVOC, D_MOD, 0, 0);
  }
}
